// GraphAttention_45397804319021
// MI455X (gfx1250) — compile-verified, matrix operand order checked by reading, not run on hardware
//
#include <hip/hip_runtime.h>
#include <hip/hip_bf16.h>
#include <stdint.h>

// ---------------------------------------------------------------------------
// GraphAttention layer for MI455X (gfx1250, wave32, WMMA).
// All GEMMs -> v_wmma_f32_16x16x32_bf16 (f32 accumulate), fully unrolled
// K-loop with prefetched fragments; epilogue fully specialized at compile
// time (ACT / BIAS / RESID template params -> branch-free stores).
// Edge softmax -> 3 memory passes with f32 atomics (HBM-bound part).
// ---------------------------------------------------------------------------

#define N_NODES 50000
#define N_EDGES 400000
#define DCH     128

typedef unsigned short u16;
typedef unsigned int   u32;
typedef __attribute__((ext_vector_type(16))) __bf16 v16bf;
typedef __attribute__((ext_vector_type(8)))  float  v8f;
typedef __attribute__((ext_vector_type(16))) u16    v16u;
typedef __attribute__((ext_vector_type(8)))  u16    v8u;

__device__ __forceinline__ u16 f2bf(float f) {
  u32 u = __builtin_bit_cast(u32, f);
  u32 r = 0x7FFFu + ((u >> 16) & 1u);   // round-to-nearest-even
  return (u16)((u + r) >> 16);
}

// A fragment: 16x32 bf16 tile from LDS, row-major with row stride sa_stride.
// ISA layout: lanes 0-15 (M=lane): K chunks [0..7],[16..23]; lanes 16-31: [8..15],[24..31].
__device__ __forceinline__ v16bf load_frag_a(const u16* sA, int sa_stride, int lane, int k0) {
  int m  = lane & 15;
  int kb = (lane >> 4) << 3;            // 0 or 8
  const u16* p = sA + m * sa_stride + k0 + kb;
  v8u lo = *(const v8u*)(p);
  v8u hi = *(const v8u*)(p + 16);
  v16u a = __builtin_shufflevector(lo, hi, 0,1,2,3,4,5,6,7,8,9,10,11,12,13,14,15);
  return __builtin_bit_cast(v16bf, a);
}

// B fragment: 32x16 bf16, weights pre-transposed in global as Wt[N][K] (bf16).
__device__ __forceinline__ v16bf load_frag_b(const u16* __restrict__ Wt, int K, int n0,
                                             int lane, int k0) {
  int n  = lane & 15;
  int kb = (lane >> 4) << 3;
  const u16* p = Wt + (size_t)(n0 + n) * K + k0 + kb;
  v8u lo = *(const v8u*)(p);
  v8u hi = *(const v8u*)(p + 16);
  v16u b = __builtin_shufflevector(lo, hi, 0,1,2,3,4,5,6,7,8,9,10,11,12,13,14,15);
  return __builtin_bit_cast(v16bf, b);
}

// --------------------------- weight transpose+cvt ---------------------------
// W[K][N] f32 row-major  ->  Wt[N][K] bf16
__global__ __launch_bounds__(256)
void ga_w_transpose(const float* __restrict__ W, u16* __restrict__ Wt, int K, int N) {
  int idx = blockIdx.x * blockDim.x + threadIdx.x;
  if (idx >= K * N) return;
  int n = idx / K;
  int k = idx - n * K;
  Wt[idx] = f2bf(W[(size_t)k * N + n]);
}

// ------------------------------- fused GEMM --------------------------------
// C[M][N] = act(A[M][K] @ Wt^T [+ bias]) [+ resid]
// ACT: 0=none 1=lrelu 2=relu  (compile-time -> branch-free epilogue)
// One 16-row tile per block; 8 waves cover N/16 column tiles.
template <int K, int ACT, bool BIAS, bool RESID>
__global__ __launch_bounds__(256)
void ga_gemm(const float* __restrict__ A,
             const u16* __restrict__ Wt, int N,
             const float* __restrict__ bias,
             const float* __restrict__ resid,
             float* __restrict__ C) {
  constexpr int SA_STRIDE = K + 8;
  constexpr int KS = K / 32;
  __shared__ u16 sA[16 * SA_STRIDE];
  const int row0 = blockIdx.x << 4;

  // stage & convert the 16xK A-tile (coalesced)
  for (int idx = threadIdx.x; idx < 16 * K; idx += blockDim.x) {
    int r = idx / K;
    int c = idx - r * K;
    sA[r * SA_STRIDE + c] = f2bf(A[(size_t)(row0 + r) * K + c]);
  }
  __syncthreads();

  const int lane   = threadIdx.x & 31;
  const int wave   = threadIdx.x >> 5;
  const int nwaves = blockDim.x >> 5;
  const int ntiles = N >> 4;

  // A fragments are shared across all column tiles of this wave: load once.
  v16bf af[KS];
#pragma unroll
  for (int ks = 0; ks < KS; ++ks) af[ks] = load_frag_a(sA, SA_STRIDE, lane, ks << 5);

  for (int nt = wave; nt < ntiles; nt += nwaves) {
    const int n0 = nt << 4;
    v16bf bf[KS];
#pragma unroll
    for (int ks = 0; ks < KS; ++ks) bf[ks] = load_frag_b(Wt, K, n0, lane, ks << 5);
    v8f acc = {};
#pragma unroll
    for (int ks = 0; ks < KS; ++ks)
      acc = __builtin_amdgcn_wmma_f32_16x16x32_bf16(false, af[ks], false, bf[ks],
                                                    (short)0, acc, false, false);
    // C/D layout: VGPR r -> (M = r + 8*(lane>=16), N = lane&15)
    const int n    = lane & 15;
    const int moff = (lane >> 4) << 3;
    const int col  = n0 + n;
    float bv = 0.0f;
    if constexpr (BIAS) bv = bias[col];
#pragma unroll
    for (int r = 0; r < 8; ++r) {
      const int row = row0 + moff + r;
      float v = acc[r];
      if constexpr (BIAS) v += bv;
      if constexpr (ACT == 1) v = (v > 0.0f) ? v : 0.01f * v;   // leaky_relu
      if constexpr (ACT == 2) v = fmaxf(v, 0.0f);               // relu
      if constexpr (RESID)    v += resid[(size_t)row * N + col];
      C[(size_t)row * N + col] = v;
    }
  }
}

// ------------------------------ edge kernels -------------------------------
__global__ __launch_bounds__(256)
void ga_init(u32* __restrict__ Mkey, float* __restrict__ S, float* __restrict__ AGG) {
  size_t idx = (size_t)blockIdx.x * blockDim.x + threadIdx.x;
  if (idx >= (size_t)N_NODES * DCH) return;
  Mkey[idx] = 0u;           // ordered-key identity for max
  S[idx]    = 0.0f;
  AGG[idx]  = 0.0f;
}

__device__ __forceinline__ u32 f32_to_okey(float f) {
  u32 u = __builtin_bit_cast(u32, f);
  return (u & 0x80000000u) ? ~u : (u | 0x80000000u);
}
__device__ __forceinline__ float okey_to_f32(u32 k) {
  u32 u = (k & 0x80000000u) ? (k & 0x7FFFFFFFu) : ~k;
  return __builtin_bit_cast(float, u);
}

// 128 consecutive threads handle one edge (channels 0..127) => the edge id is
// wave-uniform; force it into SGPRs so gather bases are scalar.
__device__ __forceinline__ int uniform_load(const int* __restrict__ p, int e) {
  return __builtin_amdgcn_readfirstlane(p[e]);
}

// pass 1: logits t = u[src]*(v[dst]+x3)/sqrt(32); segment-max via ordered-uint atomicMax
__global__ __launch_bounds__(256)
void ga_attn_logits(const float* __restrict__ U, const float* __restrict__ V,
                    const float* __restrict__ X3,
                    const int* __restrict__ src, const int* __restrict__ dst,
                    float* __restrict__ T, u32* __restrict__ Mkey) {
  size_t idx = (size_t)blockIdx.x * blockDim.x + threadIdx.x;
  if (idx >= (size_t)N_EDGES * DCH) return;
  int d = (int)(idx & (DCH - 1));
  int e = (int)(idx >> 7);
  int s  = uniform_load(src, e);
  int dd = uniform_load(dst, e);
  float t = U[(size_t)s * DCH + d] * (V[(size_t)dd * DCH + d] + X3[idx])
            * 0.17677669529663687f;  // 1/sqrt(32)
  T[idx] = t;
  atomicMax(&Mkey[(size_t)dd * DCH + d], f32_to_okey(t));
}

// pass 2: e = exp(t - max); segment-sum
__global__ __launch_bounds__(256)
void ga_attn_exp(const u32* __restrict__ Mkey, const int* __restrict__ dst,
                 float* __restrict__ T, float* __restrict__ S) {
  size_t idx = (size_t)blockIdx.x * blockDim.x + threadIdx.x;
  if (idx >= (size_t)N_EDGES * DCH) return;
  int d = (int)(idx & (DCH - 1));
  int e = (int)(idx >> 7);
  int dd = uniform_load(dst, e);
  float m  = okey_to_f32(Mkey[(size_t)dd * DCH + d]);
  float ex = __expf(T[idx] - m);
  T[idx] = ex;
  atomicAdd(&S[(size_t)dd * DCH + d], ex);
}

// pass 3: msg = attn * (k[dst] + x5); scatter-add into agg
__global__ __launch_bounds__(256)
void ga_attn_scatter(const float* __restrict__ T, const float* __restrict__ S,
                     const float* __restrict__ K4, const float* __restrict__ X5,
                     const int* __restrict__ dst, float* __restrict__ AGG) {
  size_t idx = (size_t)blockIdx.x * blockDim.x + threadIdx.x;
  if (idx >= (size_t)N_EDGES * DCH) return;
  int d = (int)(idx & (DCH - 1));
  int e = (int)(idx >> 7);
  int dd = uniform_load(dst, e);
  float a   = T[idx] / (S[(size_t)dd * DCH + d] + 1e-20f);
  float msg = a * (K4[(size_t)dd * DCH + d] + X5[idx]);
  atomicAdd(&AGG[(size_t)dd * DCH + d], msg);
}

// ----------------------------- add + layernorm -----------------------------
// one wave32 per row (128 ch, 4 per lane), shfl_xor reductions
template <bool ADD>
__global__ __launch_bounds__(256)
void ga_add_ln(const float* __restrict__ X, const float* __restrict__ Y,
               const float* __restrict__ gamma, const float* __restrict__ beta,
               float* __restrict__ out, int M) {
  int lane = threadIdx.x & 31;
  int wave = threadIdx.x >> 5;
  int row  = blockIdx.x * (blockDim.x >> 5) + wave;
  if (row >= M) return;
  const size_t base = (size_t)row * DCH;
  float v[4];
  float s = 0.0f;
#pragma unroll
  for (int i = 0; i < 4; ++i) {
    int c = lane + (i << 5);
    float t = X[base + c];
    if constexpr (ADD) t += Y[base + c];
    v[i] = t;
    s += t;
  }
#pragma unroll
  for (int off = 16; off > 0; off >>= 1) s += __shfl_xor(s, off, 32);
  float mu = s * (1.0f / DCH);
  float q = 0.0f;
#pragma unroll
  for (int i = 0; i < 4; ++i) { float dlt = v[i] - mu; q += dlt * dlt; }
#pragma unroll
  for (int off = 16; off > 0; off >>= 1) q += __shfl_xor(q, off, 32);
  float rstd = rsqrtf(q * (1.0f / DCH) + 1e-5f);
#pragma unroll
  for (int i = 0; i < 4; ++i) {
    int c = lane + (i << 5);
    out[base + c] = (v[i] - mu) * rstd * gamma[c] + beta[c];
  }
}

// --------------------------------- driver ----------------------------------
extern "C" void kernel_launch(void* const* d_in, const int* in_sizes, int n_in,
                              void* d_out, int out_size, void* d_ws, size_t ws_size,
                              hipStream_t stream) {
  const float* h      = (const float*)d_in[0];
  const float* e_ij   = (const float*)d_in[1];
  const int*   src    = (const int*)d_in[2];
  const int*   dst    = (const int*)d_in[3];
  const float* w1     = (const float*)d_in[4];
  const float* w2     = (const float*)d_in[5];
  const float* w3     = (const float*)d_in[6];
  const float* w4     = (const float*)d_in[7];
  const float* w5     = (const float*)d_in[8];
  const float* w6     = (const float*)d_in[9];
  const float* mlp_w1 = (const float*)d_in[10];
  const float* mlp_b1 = (const float*)d_in[11];
  const float* mlp_w2 = (const float*)d_in[12];
  const float* mlp_b2 = (const float*)d_in[13];
  const float* gamma  = (const float*)d_in[14];
  const float* beta   = (const float*)d_in[15];
  float* out = (float*)d_out;

  // workspace carve-up (256B aligned)
  char* ws = (char*)d_ws;
  size_t off = 0;
  auto carve = [&](size_t bytes) -> char* {
    char* p = ws + off;
    off = (off + bytes + 255) & ~(size_t)255;
    return p;
  };
  const size_t ND = (size_t)N_NODES * DCH * sizeof(float);
  const size_t ED = (size_t)N_EDGES * DCH * sizeof(float);
  u16* W1T = (u16*)carve(128 * 128 * 2);
  u16* W2T = (u16*)carve(128 * 128 * 2);
  u16* W3T = (u16*)carve(128 * 128 * 2);
  u16* W4T = (u16*)carve(128 * 128 * 2);
  u16* W5T = (u16*)carve(128 * 128 * 2);
  u16* W6T = (u16*)carve(128 * 128 * 2);
  u16* M1T = (u16*)carve(256 * 128 * 2);   // [N=256][K=128]
  u16* M2T = (u16*)carve(128 * 256 * 2);   // [N=128][K=256]
  float* U    = (float*)carve(ND);
  float* V    = (float*)carve(ND);
  float* K4   = (float*)carve(ND);
  float* H6   = (float*)carve(ND);
  float* AGG  = (float*)carve(ND);
  float* S    = (float*)carve(ND);
  float* OUT0 = (float*)carve(ND);
  float* Yb   = (float*)carve(ND);
  u32*   Mkey = (u32*)carve((size_t)N_NODES * DCH * sizeof(u32));
  float* X3   = (float*)carve(ED);
  float* X5   = (float*)carve(ED);
  float* T    = (float*)carve(ED);
  float* HID  = (float*)carve((size_t)N_NODES * 256 * sizeof(float));
  (void)ws_size; (void)in_sizes; (void)n_in; (void)out_size;

  // 1) convert + transpose weights to bf16 [N][K]
  ga_w_transpose<<<64, 256, 0, stream>>>(w1, W1T, 128, 128);
  ga_w_transpose<<<64, 256, 0, stream>>>(w2, W2T, 128, 128);
  ga_w_transpose<<<64, 256, 0, stream>>>(w3, W3T, 128, 128);
  ga_w_transpose<<<64, 256, 0, stream>>>(w4, W4T, 128, 128);
  ga_w_transpose<<<64, 256, 0, stream>>>(w5, W5T, 128, 128);
  ga_w_transpose<<<64, 256, 0, stream>>>(w6, W6T, 128, 128);
  ga_w_transpose<<<128, 256, 0, stream>>>(mlp_w1, M1T, 128, 256);
  ga_w_transpose<<<128, 256, 0, stream>>>(mlp_w2, M2T, 256, 128);

  // 2) node GEMMs (WMMA): U,V,K = lrelu(h@w*), H6 = h@w6
  const int NT = N_NODES / 16;   // 3125
  const int ET = N_EDGES / 16;   // 25000
  ga_gemm<128, 1, false, false><<<NT, 256, 0, stream>>>(h, W1T, 128, nullptr, nullptr, U);
  ga_gemm<128, 1, false, false><<<NT, 256, 0, stream>>>(h, W2T, 128, nullptr, nullptr, V);
  ga_gemm<128, 1, false, false><<<NT, 256, 0, stream>>>(h, W4T, 128, nullptr, nullptr, K4);
  ga_gemm<128, 0, false, false><<<NT, 256, 0, stream>>>(h, W6T, 128, nullptr, nullptr, H6);

  // 3) edge GEMMs (WMMA): X3 = e@w3, X5 = lrelu(e@w5)
  ga_gemm<128, 0, false, false><<<ET, 256, 0, stream>>>(e_ij, W3T, 128, nullptr, nullptr, X3);
  ga_gemm<128, 1, false, false><<<ET, 256, 0, stream>>>(e_ij, W5T, 128, nullptr, nullptr, X5);

  // 4) edge softmax + aggregate (3 passes, f32 atomics; node tables L2-resident)
  const int nodeElems = (N_NODES * DCH + 255) / 256;
  const int edgeElems = (int)(((size_t)N_EDGES * DCH + 255) / 256);
  ga_init<<<nodeElems, 256, 0, stream>>>(Mkey, S, AGG);
  ga_attn_logits<<<edgeElems, 256, 0, stream>>>(U, V, X3, src, dst, T, Mkey);
  ga_attn_exp<<<edgeElems, 256, 0, stream>>>(Mkey, dst, T, S);
  ga_attn_scatter<<<edgeElems, 256, 0, stream>>>(T, S, K4, X5, dst, AGG);

  // 5) out0 = LN(h@w6 + agg)
  ga_add_ln<true><<<(N_NODES + 7) / 8, 256, 0, stream>>>(H6, AGG, gamma, beta, OUT0, N_NODES);

  // 6) MLP (WMMA): hid = relu(out0@mlp_w1 + b1); y = out0 + hid@mlp_w2 + b2
  ga_gemm<128, 2, true, false><<<NT, 256, 0, stream>>>(OUT0, M1T, 256, mlp_b1, nullptr, HID);
  ga_gemm<256, 0, true, true><<<NT, 256, 0, stream>>>(HID, M2T, 128, mlp_b2, OUT0, Yb);

  // 7) final layernorm -> d_out
  ga_add_ln<false><<<(N_NODES + 7) / 8, 256, 0, stream>>>(Yb, nullptr, gamma, beta, out, N_NODES);
}
